// mRNN_86715389706694
// MI455X (gfx1250) — compile-verified, matrix operand order checked by reading, not run on hardware
//
#include <hip/hip_runtime.h>
#include <hip/hip_bf16.h>

#define H       1612
#define HK      1632          // K padded to multiple of 32
#define NTILES  101           // ceil(1612/16) N tiles (N padded to 1616)
#define KTILES  51            // 1632/32 K tiles
#define TSTEPS  500
#define NB      32
#define ALM_LO  1280
#define ALM_HI  1460

#define STRIP_BYTES  (KTILES * 512 * 2)       // 52,224 B per N-strip
#define STRIP_VECS   (STRIP_BYTES / 16)       // 3,264 x b128

typedef __attribute__((ext_vector_type(16))) __bf16 v16bf;
typedef __attribute__((ext_vector_type(8)))  float  v8f;

// ---------------------------------------------------------------------------
// Kernel 1: build W_eff^T in bf16, pre-swizzled into WMMA-B fragment tiles.
// B[k][n] = (mask[n,k]*clip(W_rec[n,k],0,1) + fixed[n,k]) * sign[k,k]
// Tile layout: [ntile][ktile][lane*16 + i], i -> k per the 16-bit operand map:
//   lane<16 : k_local = {0..7, 16..23}[i] ;  lane>=16 : k_local = {8..15, 24..31}[i]
// ---------------------------------------------------------------------------
__global__ void pack_w_kernel(const float* __restrict__ W_rec,
                              const float* __restrict__ Wmask,
                              const float* __restrict__ Wfix,
                              const float* __restrict__ Wsign,
                              __bf16* __restrict__ Wt) {
  int tile = blockIdx.x;                 // 0 .. NTILES*KTILES-1
  int nt = tile / KTILES, kt = tile % KTILES;
  int tid  = threadIdx.x;                // 0..511
  int lane = tid >> 4, i = tid & 15;
  int half = lane >> 4;
  int kl   = (i < 8) ? (i + (half ? 8 : 0)) : (i - 8 + (half ? 24 : 16));
  int n = nt * 16 + (lane & 15);
  int k = kt * 32 + kl;
  float v = 0.0f;
  if (n < H && k < H) {
    float w = W_rec[(size_t)n * H + k];
    w = fminf(fmaxf(w, 0.0f), 1.0f);
    v = (Wmask[(size_t)n * H + k] * w + Wfix[(size_t)n * H + k]) *
        Wsign[(size_t)k * H + k];
  }
  Wt[(size_t)tile * 512 + tid] = (__bf16)v;
}

// ---------------------------------------------------------------------------
// Kernel 2: initialize state, output buffer and grid-barrier counter.
// ---------------------------------------------------------------------------
__global__ void init_state_kernel(const float* __restrict__ hn,
                                  const float* __restrict__ xn,
                                  float* __restrict__ x,
                                  __bf16* __restrict__ hA,
                                  __bf16* __restrict__ hB,
                                  float* __restrict__ out,
                                  unsigned long long* __restrict__ bar) {
  int idx = blockIdx.x * blockDim.x + threadIdx.x;
  if (idx == 0) *bar = 0ull;
  if (idx < NB * HK) {
    int b = idx / HK, k = idx % HK;
    float xv = (k < H) ? xn[(size_t)b * H + k] : 0.0f;
    float hv = (k < H) ? hn[(size_t)b * H + k] : 0.0f;
    x[idx]  = xv;
    hA[idx] = (__bf16)hv;
    hB[idx] = (__bf16)0.0f;
  }
  if (idx < NB * TSTEPS) out[idx] = 0.0f;
}

// ---------------------------------------------------------------------------
// Kernel 3: persistent scan. 26 blocks x 256 threads (8 waves).
// Wave w of block g owns (ntile = 4g + (w>>1), mtile = w&1).
// B strips staged once into LDS via global_load_async_to_lds_b128 (ASYNCcnt).
// Grid barrier between steps via monotonic agent-scope atomic counter.
// ---------------------------------------------------------------------------
__global__ __launch_bounds__(256, 1) void rnn_scan_kernel(
    const __bf16* __restrict__ Wt,
    float* __restrict__ x,
    __bf16* __restrict__ hA, __bf16* __restrict__ hB,
    const float* __restrict__ inp, const float* __restrict__ cue,
    const float* __restrict__ inhib,
    const float* __restrict__ W_inp, const float* __restrict__ W_cue,
    const float* __restrict__ outw, const float* __restrict__ basef,
    float* __restrict__ out,
    unsigned long long* __restrict__ bar) {
  extern __shared__ char smem_raw[];
  __bf16* sW = (__bf16*)smem_raw;

  const int g = blockIdx.x;

  // ---- one-shot async staging of this block's B-strips into LDS ----
  // global_load_async_to_lds_b128: VDST = LDS byte address, VADDR = 64-bit
  // global address (GV mode). Tracked by ASYNCcnt.
  {
    const int strips = (NTILES - g * 4) < 4 ? (NTILES - g * 4) : 4;
    const int totalv = strips * STRIP_VECS;                 // b128 transfers
    const unsigned long long gbase =
        (unsigned long long)(uintptr_t)Wt +
        (unsigned long long)g * 4 * STRIP_BYTES;
    const unsigned int lbase = (unsigned int)(uintptr_t)(void*)smem_raw;
    for (int e = threadIdx.x; e < totalv; e += 256) {
      unsigned int       loff = lbase + (unsigned int)e * 16u;
      unsigned long long ga   = gbase + (unsigned long long)e * 16ull;
      asm volatile("global_load_async_to_lds_b128 %0, %1, off"
                   :: "v"(loff), "v"(ga)
                   : "memory");
    }
    asm volatile("s_wait_asynccnt 0x0" ::: "memory");
  }
  __syncthreads();

  const int wave  = threadIdx.x >> 5;
  const int lane  = threadIdx.x & 31;
  const int strip = wave >> 1;
  const int mtile = wave & 1;
  const int nt    = g * 4 + strip;
  const bool active = (nt < NTILES);
  const int half  = lane >> 4;
  const int nloc  = lane & 15;
  const int n     = nt * 16 + nloc;

  float winp_n = 0.f, wcue_n = 0.f, bf_n = 0.f, outw_n = 0.f;
  if (active && n < H) {
    winp_n = W_inp[n];
    wcue_n = W_cue[n];
    bf_n   = basef[n];
    if (n >= ALM_LO && n < ALM_HI) outw_n = outw[n - ALM_LO];
  }
  const bool hasout = active && nt >= (ALM_LO / 16) && nt <= ((ALM_HI - 1) / 16);

  const int arow = mtile * 16 + nloc;                          // A-matrix row
  const __bf16* sWb = sW + (size_t)strip * KTILES * 512 + (size_t)lane * 16;

  for (int t = 0; t < TSTEPS; ++t) {
    const __bf16* hc  = (t & 1) ? hB : hA;
    __bf16*       hnx = (t & 1) ? hA : hB;

    if (active) {
      v8f acc = {0.f, 0.f, 0.f, 0.f, 0.f, 0.f, 0.f, 0.f};
      const __bf16* ap = hc + (size_t)arow * HK + (half ? 8 : 0);
      for (int kt = 0; kt < KTILES; ++kt) {
        union { v16bf v; uint4 u[2]; } a, b;
        a.u[0] = *(const uint4*)(ap + kt * 32);        // k chunk {0..7}
        a.u[1] = *(const uint4*)(ap + kt * 32 + 16);   // k chunk {16..23}
        b.u[0] = *(const uint4*)(sWb + (size_t)kt * 512);
        b.u[1] = *(const uint4*)(sWb + (size_t)kt * 512 + 8);
        acc = __builtin_amdgcn_wmma_f32_16x16x32_bf16(
            false, a.v, false, b.v, (short)0, acc, false, false);
      }

      // epilogue: leaky integrate + ReLU + output projection (fully unrolled
      // so acc[r] is a static extract, no v_movrels)
#pragma unroll
      for (int r = 0; r < 8; ++r) {
        float contrib = 0.0f;
        const int bidx = mtile * 16 + r + (half ? 8 : 0);  // C layout: M = r (+8)
        const int bt = bidx * TSTEPS + t;
        if (n < H) {
          float ff = acc[r] + inp[bt] * winp_n + cue[bt] * wcue_n +
                     inhib[(size_t)bt * H + n];
          // hide next step's HBM-streamed inhib element behind the barrier
          if (t + 1 < TSTEPS)
            __builtin_prefetch(&inhib[(size_t)(bt + 1) * H + n], 0, 3);
          float xo = x[(size_t)bidx * HK + n];
          float xnw = xo + 0.1f * (ff - xo);
          float hv = xnw + bf_n;
          hv = hv > 0.0f ? hv : 0.0f;
          x[(size_t)bidx * HK + n] = xnw;
          hnx[(size_t)bidx * HK + n] = (__bf16)hv;
          contrib = hv * outw_n;
        }
        if (hasout) {
          // reduce the 16 lanes that share this (b,t)
          for (int o = 8; o > 0; o >>= 1) contrib += __shfl_xor(contrib, o, 16);
          if (nloc == 0) atomicAdd(&out[bt], contrib);
        }
      }
    }

    // ---- grid-wide step barrier (monotonic counter, agent scope) ----
    __threadfence();
    __syncthreads();
    if (threadIdx.x == 0) {
      __hip_atomic_fetch_add(bar, 1ull, __ATOMIC_RELEASE,
                             __HIP_MEMORY_SCOPE_AGENT);
      const unsigned long long target =
          (unsigned long long)(t + 1) * gridDim.x;
      while (__hip_atomic_load(bar, __ATOMIC_ACQUIRE,
                               __HIP_MEMORY_SCOPE_AGENT) < target)
        __builtin_amdgcn_s_sleep(2);
    }
    __syncthreads();
    __threadfence();
  }
}

// ---------------------------------------------------------------------------
extern "C" void kernel_launch(void* const* d_in, const int* in_sizes, int n_in,
                              void* d_out, int out_size, void* d_ws, size_t ws_size,
                              hipStream_t stream) {
  const float* inp   = (const float*)d_in[0];   // [B,T,1]
  const float* cue   = (const float*)d_in[1];   // [B,T,1]
  const float* hn    = (const float*)d_in[2];   // [1,B,H]
  const float* xn    = (const float*)d_in[3];   // [1,B,H]
  const float* inhib = (const float*)d_in[4];   // [B,T,H]
  const float* W_rec = (const float*)d_in[5];   // [H,H]
  const float* Wmask = (const float*)d_in[6];   // [H,H]
  const float* Wfix  = (const float*)d_in[7];   // [H,H]
  const float* Wsign = (const float*)d_in[8];   // [H,H] (diag)
  const float* W_inp = (const float*)d_in[9];   // [H,1]
  const float* W_cue = (const float*)d_in[10];  // [H,1]
  const float* outw  = (const float*)d_in[11];  // [1,180]
  const float* basef = (const float*)d_in[12];  // [H]
  float* out = (float*)d_out;                   // [B,T,1]

  char* ws = (char*)d_ws;
  unsigned long long* bar = (unsigned long long*)ws;
  __bf16* Wt = (__bf16*)(ws + 256);
  const size_t wt_bytes = (size_t)NTILES * STRIP_BYTES;        // 5,274,624 B
  float*  x  = (float*)(ws + 256 + wt_bytes);
  __bf16* hA = (__bf16*)((char*)x + (size_t)NB * HK * sizeof(float));
  __bf16* hB = hA + (size_t)NB * HK;

  pack_w_kernel<<<NTILES * KTILES, 512, 0, stream>>>(W_rec, Wmask, Wfix, Wsign, Wt);
  init_state_kernel<<<(NB * HK + 255) / 256, 256, 0, stream>>>(hn, xn, x, hA, hB,
                                                               out, bar);

  const size_t shmem = (size_t)4 * STRIP_BYTES;  // 208,896 B of WGP LDS
  hipFuncSetAttribute((const void*)rnn_scan_kernel,
                      hipFuncAttributeMaxDynamicSharedMemorySize, (int)shmem);
  rnn_scan_kernel<<<26, 256, shmem, stream>>>(Wt, x, hA, hB, inp, cue, inhib,
                                              W_inp, W_cue, outw, basef, out, bar);
}
